// NTXent_11132555231335
// MI455X (gfx1250) — compile-verified, matrix operand order checked by reading, not run on hardware
//
#include <hip/hip_runtime.h>

// ---------------------------------------------------------------------------
// NT-Xent loss, fused for MI455X (gfx1250, wave32, WMMA f16 16x16x32).
//   z = [z_i; z_j]  (2N x D),  zn = row-normalized
//   sim = zn @ zn^T   (8192 x 8192, never materialized, SYMMETRIC)
//   rowsum[r] = sum_c exp(sim[r][c] / t)   (fused epilogue; lower-triangular
//               block grid only -- off-diag tiles feed both row & col sums)
//   pos[i] = pos[i+N] = dot(zn_i, zn_{i+N})    (exact f32)
//   loss = mean_r ( -(pos[r]/t - log(rowsum[r]/2N)) )
// ---------------------------------------------------------------------------

#define N_PAIRS   4096
#define TWO_N     8192
#define DIM       256
#define INV_T     10.0f
#define EPS_N     1e-8f

typedef __attribute__((ext_vector_type(16))) _Float16 v16h;
typedef __attribute__((ext_vector_type(8)))  float    v8f;

union Frag16 {           // one WMMA 16-bit A/B operand: 8 VGPRs = 16 halves
    v16h h;
    uint4 u[2];
};

// ---------------------------------------------------------------------------
// Kernel 1: row L2 norms -> f16 normalized copy; zero row-sum accumulators.
// One 256-thread block per row (D == 256).
// ---------------------------------------------------------------------------
__global__ void ntx_normalize(const float* __restrict__ zi,
                              const float* __restrict__ zj,
                              _Float16* __restrict__ zn,
                              float* __restrict__ norms,
                              float* __restrict__ rowsum) {
    const int r = blockIdx.x;
    const int t = threadIdx.x;
    const float* src = (r < N_PAIRS) ? (zi + (size_t)r * DIM)
                                     : (zj + (size_t)(r - N_PAIRS) * DIM);
    float v = src[t];
    float ss = v * v;
    ss += __shfl_xor(ss, 16, 32);
    ss += __shfl_xor(ss,  8, 32);
    ss += __shfl_xor(ss,  4, 32);
    ss += __shfl_xor(ss,  2, 32);
    ss += __shfl_xor(ss,  1, 32);
    __shared__ float red[8];
    __shared__ float s_inv;
    if ((t & 31) == 0) red[t >> 5] = ss;
    __syncthreads();
    if (t == 0) {
        float tot = 0.f;
        #pragma unroll
        for (int i = 0; i < 8; ++i) tot += red[i];
        float nrm = sqrtf(tot);
        norms[r]  = nrm;
        rowsum[r] = 0.f;                       // re-zeroed every launch
        s_inv = 1.0f / fmaxf(nrm, EPS_N);
    }
    __syncthreads();
    zn[(size_t)r * DIM + t] = (_Float16)(v * s_inv);
}

// ---------------------------------------------------------------------------
// Kernel 2: fused sim-GEMM + exp + row-sum, lower-triangular block grid.
// Block = 256 threads = 8 waves, block tile 128x128 at (by, bx), by >= bx.
// Wave (wm in 0..3, wn in 0..1) computes a 32x64 strip = 2x4 subtiles of
// 16x16 via v_wmma_f32_16x16x32_f16, K-loop over DIM in steps of 32.
// Operands come straight from global (4 MB zn lives entirely in 192 MB L2).
//
// __launch_bounds__(256, 1): the steady-state live set (64 acc VGPRs +
// 48 VGPRs of in-flight b128 load data + pointers) needs ~130+ VGPRs;
// letting the allocator take them eliminates the accumulator spills seen
// at the default occupancy target. The kernel is latency-tolerant (clause-
// grouped L2 loads overlapped with long WMMA chains), so low occupancy is
// the right trade on MI455X.
//
// 16-bit A-frag layout (ISA 7.12.2): lane L<16 -> row M=L, halves 0..7 hold
// K=k0..k0+7, halves 8..15 hold K=k0+16..k0+23; lanes 16..31 same rows with
// K offset +8.  B = zn^T, so B-frags use identical addressing keyed by col.
//
// Epilogue: e = exp(sim/t), accumulated into
//   - row partials  (reduce over the 16 lanes of each half; M = j + half*8)
//   - col partials  (reduce over the 8 VGPRs + the two halves via xor-16)
// Off-diagonal blocks scatter col partials into rowsum[col] by symmetry.
// ---------------------------------------------------------------------------
__global__ void __launch_bounds__(256, 1)
ntx_simexp(const _Float16* __restrict__ zn,
           float* __restrict__ rowsum) {
    const int bx = blockIdx.x;              // column block
    const int by = blockIdx.y;              // row block
    if (by < bx) return;                    // uniform: lower triangle only

    const int lane = threadIdx.x & 31;
    const int wave = threadIdx.x >> 5;      // 0..7
    const int half = lane >> 4;             // 0/1
    const int lr   = lane & 15;             // 0..15

    const int wm = wave & 3;                // 4 row strips of 32
    const int wn = wave >> 2;               // 2 col strips of 64

    const int row0 = by * 128 + wm * 32;
    const int col0 = bx * 128 + wn * 64;

    // per-lane fragment base pointers (uint4 = 8 halves)
    const uint4* aptr[2];
    #pragma unroll
    for (int mi = 0; mi < 2; ++mi)
        aptr[mi] = (const uint4*)(zn + (size_t)(row0 + mi * 16 + lr) * DIM + half * 8);
    const uint4* bptr[4];
    #pragma unroll
    for (int ni = 0; ni < 4; ++ni)
        bptr[ni] = (const uint4*)(zn + (size_t)(col0 + ni * 16 + lr) * DIM + half * 8);

    v8f acc[2][4] = {};

    #pragma unroll
    for (int k0 = 0; k0 < DIM; k0 += 32) {
        const int q = k0 >> 3;              // uint4 index of K chunk
        Frag16 a[2], b[4];
        #pragma unroll
        for (int mi = 0; mi < 2; ++mi) {
            a[mi].u[0] = aptr[mi][q];       // K = k0 + half*8 + 0..7
            a[mi].u[1] = aptr[mi][q + 2];   // K = k0 + 16 + half*8 + 0..7
        }
        #pragma unroll
        for (int ni = 0; ni < 4; ++ni) {
            b[ni].u[0] = bptr[ni][q];
            b[ni].u[1] = bptr[ni][q + 2];
        }
        #pragma unroll
        for (int mi = 0; mi < 2; ++mi)
            #pragma unroll
            for (int ni = 0; ni < 4; ++ni)
                acc[mi][ni] = __builtin_amdgcn_wmma_f32_16x16x32_f16(
                    /*neg_a=*/false, a[mi].h,
                    /*neg_b=*/false, b[ni].h,
                    /*c_mod=*/(short)0, acc[mi][ni],
                    /*reuse_a=*/false, /*reuse_b=*/false);
    }

    // ---- epilogue ---------------------------------------------------------
    // Overwrite acc elements with exp(sim/t) in place, then reduce.
    #pragma unroll
    for (int mi = 0; mi < 2; ++mi)
        #pragma unroll
        for (int ni = 0; ni < 4; ++ni)
            #pragma unroll
            for (int j = 0; j < 8; ++j)
                acc[mi][ni][j] = __expf(acc[mi][ni][j] * INV_T);

    // row sums: sum over ni, then reduce across the 16 lanes of each half
    #pragma unroll
    for (int mi = 0; mi < 2; ++mi)
        #pragma unroll
        for (int j = 0; j < 8; ++j) {
            float x = acc[mi][0][j] + acc[mi][1][j] + acc[mi][2][j] + acc[mi][3][j];
            x += __shfl_xor(x, 1, 32);
            x += __shfl_xor(x, 2, 32);
            x += __shfl_xor(x, 4, 32);
            x += __shfl_xor(x, 8, 32);      // stays within each 16-lane half
            if (lr == 0) {
                const int row = row0 + mi * 16 + half * 8 + j;
                atomicAdd(&rowsum[row], x);
            }
        }

    // col sums (symmetry): only for strictly off-diagonal blocks
    if (by != bx) {
        #pragma unroll
        for (int ni = 0; ni < 4; ++ni) {
            float x = 0.f;
            #pragma unroll
            for (int mi = 0; mi < 2; ++mi)
                #pragma unroll
                for (int j = 0; j < 8; ++j)
                    x += acc[mi][ni][j];    // this half's 16 rows
            x += __shfl_xor(x, 16, 32);     // combine halves -> all 32 rows
            if (half == 0) {
                const int col = col0 + ni * 16 + lr;
                atomicAdd(&rowsum[col], x);
            }
        }
    }
}

// ---------------------------------------------------------------------------
// Kernel 3: exact positives. pos[i] = pos[i+N] = dot(z_i[i], z_j[i]) /
// (max(|z_i[i]|,eps) * max(|z_j[i]|,eps)).  One block per pair.
// ---------------------------------------------------------------------------
__global__ void ntx_positives(const float* __restrict__ zi,
                              const float* __restrict__ zj,
                              const float* __restrict__ norms,
                              float* __restrict__ pos) {
    const int i = blockIdx.x;
    const int t = threadIdx.x;
    float p = zi[(size_t)i * DIM + t] * zj[(size_t)i * DIM + t];
    p += __shfl_xor(p, 16, 32);
    p += __shfl_xor(p,  8, 32);
    p += __shfl_xor(p,  4, 32);
    p += __shfl_xor(p,  2, 32);
    p += __shfl_xor(p,  1, 32);
    __shared__ float red[8];
    if ((t & 31) == 0) red[t >> 5] = p;
    __syncthreads();
    if (t == 0) {
        float tot = 0.f;
        #pragma unroll
        for (int k = 0; k < 8; ++k) tot += red[k];
        float d = fmaxf(norms[i], EPS_N) * fmaxf(norms[i + N_PAIRS], EPS_N);
        float v = tot / d;
        pos[i] = v;
        pos[i + N_PAIRS] = v;
    }
}

// ---------------------------------------------------------------------------
// Kernel 4: final scalar reduce.
// ---------------------------------------------------------------------------
__global__ void ntx_loss(const float* __restrict__ rowsum,
                         const float* __restrict__ pos,
                         float* __restrict__ out) {
    const int t = threadIdx.x;
    float acc = 0.f;
    for (int r = t; r < TWO_N; r += 256) {
        float mean = rowsum[r] * (1.0f / TWO_N);
        acc += -(pos[r] * INV_T - logf(mean));
    }
    acc += __shfl_xor(acc, 16, 32);
    acc += __shfl_xor(acc,  8, 32);
    acc += __shfl_xor(acc,  4, 32);
    acc += __shfl_xor(acc,  2, 32);
    acc += __shfl_xor(acc,  1, 32);
    __shared__ float red[8];
    if ((t & 31) == 0) red[t >> 5] = acc;
    __syncthreads();
    if (t == 0) {
        float tot = 0.f;
        #pragma unroll
        for (int k = 0; k < 8; ++k) tot += red[k];
        out[0] = tot * (1.0f / TWO_N);
    }
}

// ---------------------------------------------------------------------------
extern "C" void kernel_launch(void* const* d_in, const int* in_sizes, int n_in,
                              void* d_out, int out_size, void* d_ws, size_t ws_size,
                              hipStream_t stream) {
    const float* zi = (const float*)d_in[0];
    const float* zj = (const float*)d_in[1];
    float* out = (float*)d_out;

    char* ws = (char*)d_ws;
    _Float16* zn   = (_Float16*)ws;                               // 2N*D halves = 4 MB
    float*    nrm  = (float*)(ws + (size_t)TWO_N * DIM * 2);      // 2N
    float*    rsum = nrm + TWO_N;                                 // 2N
    float*    pos  = rsum + TWO_N;                                // 2N
    (void)in_sizes; (void)n_in; (void)out_size; (void)ws_size;

    ntx_normalize<<<TWO_N, 256, 0, stream>>>(zi, zj, zn, nrm, rsum);

    dim3 grid(TWO_N / 128, TWO_N / 128);   // lower-triangular 128x128 tiles
    ntx_simexp<<<grid, 256, 0, stream>>>(zn, rsum);

    ntx_positives<<<N_PAIRS, 256, 0, stream>>>(zi, zj, nrm, pos);

    ntx_loss<<<1, 256, 0, stream>>>(rsum, pos, out);
}
